// VQVAE_8598524526962
// MI455X (gfx1250) — matrix operand order repaired and checked by reading, not compiled, not run on hardware
//
#include <hip/hip_runtime.h>

// ---------------------------------------------------------------------------
// VQ-VAE quantization for MI455X (gfx1250, wave32, WMMA + async-to-LDS)
//   x:        [B=16, C=256, H=32, W=32] fp32
//   codebook: [K=8192, C=256] fp32
// d_out: quantized [B,C,H,W] fp32 ++ codebook_loss ++ commitment_loss
//
// Main kernel: 256 workgroups x 128 threads (4 waves). Each wave owns 16 rows
// (A frags in registers, bf16 hi/lo). All waves share double-buffered LDS
// codebook tiles staged with GLOBAL_LOAD_ASYNC_TO_LDS_B128 (fallback: sync
// copy). 3 WMMAs per 32-C step (bf16x3), fused argmin of e2[k] - 2*x.e.
// ---------------------------------------------------------------------------

typedef __attribute__((ext_vector_type(16))) __bf16 v16bf;
typedef __attribute__((ext_vector_type(8)))  float  v8f;
typedef __attribute__((ext_vector_type(4)))  int    v4i_t;

#define CDIM 256
#define KCB  8192
#define NROW 16384   // B*H*W
#define BB   16
#define HH   32
#define WW   32

#define HAVE_ASYNC_LDS __has_builtin(__builtin_amdgcn_global_load_async_to_lds_b128)

union AFrag { unsigned u[8]; v16bf v; };
union BFrag { uint4    q[2]; v16bf v; };

// XOR-swizzle 64B chunks within a 512B row so the 16 B-reader lanes spread
// across all 64 LDS banks (row stride stays 512B -> 64KB total for 2 buffers).
__device__ __forceinline__ int swz(int r, int byteOff) {
    return (((byteOff >> 6) ^ (r & 7)) << 6) | (byteOff & 63);
}

__device__ __forceinline__ void asyncCopy16(void* dst, const void* src) {
#if HAVE_ASYNC_LDS
    // Signature (from hipcc diagnostic): arg0 = v4i* in global AS, non-const.
    __builtin_amdgcn_global_load_async_to_lds_b128(
        (__attribute__((address_space(1))) v4i_t*)(v4i_t*)const_cast<void*>(src),
        (__attribute__((address_space(3))) v4i_t*)(v4i_t*)dst,
        0, 0);
#else
    *(uint4*)dst = *(const uint4*)src;     // global_load_b128 + ds_store_b128
#endif
}
__device__ __forceinline__ void asyncWait16() {
#if HAVE_ASYNC_LDS
#if __has_builtin(__builtin_amdgcn_s_wait_asynccnt)
    __builtin_amdgcn_s_wait_asynccnt(16);
#else
    asm volatile("s_wait_asynccnt 16" ::: "memory");
#endif
#endif
}
__device__ __forceinline__ void asyncWait0() {
#if HAVE_ASYNC_LDS
#if __has_builtin(__builtin_amdgcn_s_wait_asynccnt)
    __builtin_amdgcn_s_wait_asynccnt(0);
#else
    asm volatile("s_wait_asynccnt 0" ::: "memory");
#endif
#endif
}

// ---------------------------------------------------------------------------
// Kernel 1: codebook -> bf16 hi/lo + e2[k] = ||codebook[k]||^2
// ---------------------------------------------------------------------------
__global__ __launch_bounds__(256) void vq_cb_prep(const float* __restrict__ cb,
                                                  __bf16* __restrict__ cbhi,
                                                  __bf16* __restrict__ cblo,
                                                  float* __restrict__ e2) {
    const int wave = threadIdx.x >> 5;
    const int lane = threadIdx.x & 31;
    const int k    = blockIdx.x * 8 + wave;
    const float* src = cb + (size_t)k * CDIM + lane * 8;
    float4 f0 = *(const float4*)(src);
    float4 f1 = *(const float4*)(src + 4);
    float vbuf[8] = {f0.x, f0.y, f0.z, f0.w, f1.x, f1.y, f1.z, f1.w};
    float ss = 0.0f;
    __bf16* dhi = cbhi + (size_t)k * CDIM + lane * 8;
    __bf16* dlo = cblo + (size_t)k * CDIM + lane * 8;
#pragma unroll
    for (int j = 0; j < 8; ++j) {
        float v = vbuf[j];
        __bf16 hi = (__bf16)v;
        __bf16 lo = (__bf16)(v - (float)hi);
        dhi[j] = hi;
        dlo[j] = lo;
        ss = fmaf(v, v, ss);
    }
#pragma unroll
    for (int off = 16; off > 0; off >>= 1)
        ss += __shfl_xor(ss, off, 32);
    if (lane == 0) e2[k] = ss;
}

// ---------------------------------------------------------------------------
// Kernel 2: x [B,C,H,W] -> flat [N,C] bf16 hi/lo (LDS-tiled transpose)
// ---------------------------------------------------------------------------
__global__ __launch_bounds__(256) void vq_x_prep(const float* __restrict__ x,
                                                 __bf16* __restrict__ xhi,
                                                 __bf16* __restrict__ xlo) {
    __shared__ float tile[32][33];
    const int b  = blockIdx.x >> 5;
    const int h  = blockIdx.x & 31;
    const int c0 = blockIdx.y * 32;
    const int tx = threadIdx.x;
#pragma unroll
    for (int i = 0; i < 4; ++i) {
        int cy = threadIdx.y + i * 8;
        tile[cy][tx] = x[(((size_t)b * CDIM + c0 + cy) * HH + h) * WW + tx];
    }
    __syncthreads();
#pragma unroll
    for (int i = 0; i < 4; ++i) {
        int w = threadIdx.y + i * 8;
        float v = tile[tx][w];
        __bf16 hi = (__bf16)v;
        __bf16 lo = (__bf16)(v - (float)hi);
        size_t n = (size_t)b * 1024 + h * 32 + w;
        xhi[n * CDIM + c0 + tx] = hi;
        xlo[n * CDIM + c0 + tx] = lo;
    }
}

// ---------------------------------------------------------------------------
// Kernel 3: fused GEMM (bf16x3 WMMA) + argmin, LDS-staged codebook tiles.
// grid: NROW/64 = 256 blocks of 128 threads (4 waves, 16 rows per wave).
// LDS: smem[buf][hl][32 rows * 512B] = 64KB, double-buffered 2-ktile stages.
// ---------------------------------------------------------------------------
__global__ __launch_bounds__(128) void vq_argmin(const __bf16* __restrict__ xhi,
                                                 const __bf16* __restrict__ xlo,
                                                 const __bf16* __restrict__ cbhi,
                                                 const __bf16* __restrict__ cblo,
                                                 const float* __restrict__ e2,
                                                 int* __restrict__ idxOut) {
    __shared__ __attribute__((aligned(16))) char smem[2][2][32 * 512];

    const int tid     = threadIdx.x;
    const int wave    = tid >> 5;
    const int lane    = tid & 31;
    const int rowBase = blockIdx.x * 64 + wave * 16;
    const int m       = lane & 15;
    const int half    = lane >> 4;

    // ---- A fragments in registers (ISA 16-bit A 16x32 layout) --------------
    // lane-half h, VGPR v in 0..3: K = 8h + 2v ; v in 4..7: K = 16 + 8h + 2(v-4)
    AFrag ahi[8], alo[8];
    {
        const __bf16* xh = xhi + (size_t)(rowBase + m) * CDIM;
        const __bf16* xl = xlo + (size_t)(rowBase + m) * CDIM;
#pragma unroll
        for (int cs = 0; cs < 8; ++cs) {
#pragma unroll
            for (int v = 0; v < 8; ++v) {
                int koff = (v < 4) ? (8 * half + 2 * v) : (16 + 8 * half + 2 * (v - 4));
                ahi[cs].u[v] = *(const unsigned*)(xh + cs * 32 + koff);
                alo[cs].u[v] = *(const unsigned*)(xl + cs * 32 + koff);
            }
        }
    }

    float minv[8];
    int   mini[8];
#pragma unroll
    for (int r = 0; r < 8; ++r) { minv[r] = 3.4e38f; mini[r] = 0; }

    const int n   = lane & 15;        // C/D layout: N = lane%16 -> codebook col
    const int kb2 = half * 16;        // B layout: lane-half h holds K=8h..8h+7 and 16+8h..16+8h+7 (bytes)

    const char* gsrc[2] = {(const char*)cbhi, (const char*)cblo};

    // Stage 2 ktiles (32 codebook rows, hi+lo = 32KB) into smem[buf].
    // 2048 16B-chunks / 128 threads = 16 async b128 copies per thread.
    auto stageTile = [&](int buf, int ktb) {
#pragma unroll
        for (int j = 0; j < 16; ++j) {
            int c   = tid + j * 128;            // 0..2047
            int hl  = c >> 10;                  // hi block then lo block
            int r   = (c >> 5) & 31;            // local codebook row
            int off = (c & 31) << 4;            // 16B chunk within 512B row
            const char* src = gsrc[hl] + ((size_t)(ktb * 16 + r) * 512 + off);
            char* dst = &smem[buf][hl][r * 512 + swz(r, off)];
            asyncCopy16(dst, src);
        }
    };

    stageTile(0, 0);
    for (int ktb = 0; ktb < KCB / 16; ktb += 2) {
        const int cur = (ktb >> 1) & 1;
        if (ktb + 2 < KCB / 16) {
            stageTile(1 - cur, ktb + 2);
            asyncWait16();                       // prior stage's 16 done (in order)
        } else {
            asyncWait0();
        }
        __syncthreads();                         // smem[cur] visible to all waves

#pragma unroll
        for (int s = 0; s < 2; ++s) {            // the two 16x16 k-tiles staged
            const int lr = s * 16 + n;           // local codebook row for this lane
            const int k  = (ktb + s) * 16 + n;
            v8f acc = {};
#pragma unroll
            for (int cs = 0; cs < 8; ++cs) {
                const int bo = swz(lr, cs * 64 + kb2);
                BFrag bhf, blf;
                bhf.q[0] = *(const uint4*)&smem[cur][0][lr * 512 + bo];
                bhf.q[1] = *(const uint4*)&smem[cur][0][lr * 512 + swz(lr, cs * 64 + kb2 + 32)];
                blf.q[0] = *(const uint4*)&smem[cur][1][lr * 512 + bo];
                blf.q[1] = *(const uint4*)&smem[cur][1][lr * 512 + swz(lr, cs * 64 + kb2 + 32)];
                // bf16x3: hi*hi + hi*lo + lo*hi
                acc = __builtin_amdgcn_wmma_f32_16x16x32_bf16(
                    false, ahi[cs].v, false, bhf.v, (short)0, acc, false, false);
                acc = __builtin_amdgcn_wmma_f32_16x16x32_bf16(
                    false, ahi[cs].v, false, blf.v, (short)0, acc, false, false);
                acc = __builtin_amdgcn_wmma_f32_16x16x32_bf16(
                    false, alo[cs].v, false, bhf.v, (short)0, acc, false, false);
            }
            const float e2v = e2[k];
#pragma unroll
            for (int r = 0; r < 8; ++r) {
                float sc = fmaf(-2.0f, acc[r], e2v);   // x^2 const over k -> skip
                if (sc < minv[r]) { minv[r] = sc; mini[r] = k; }
            }
        }
        __syncthreads();                         // all reads of smem[cur] done
    }

    // ---- Per-wave argmin across the 16 column lanes (tie -> smaller k) -----
#pragma unroll
    for (int r = 0; r < 8; ++r) {
        float v = minv[r];
        int   i = mini[r];
#pragma unroll
        for (int off = 1; off < 16; off <<= 1) {
            float ov = __shfl_xor(v, off, 32);
            int   oi = __shfl_xor(i, off, 32);
            if (ov < v || (ov == v && oi < i)) { v = ov; i = oi; }
        }
        if ((lane & 15) == 0)
            idxOut[rowBase + half * 8 + r] = i;
    }
}

// ---------------------------------------------------------------------------
// Kernel 4: zero the two loss accumulators.
// ---------------------------------------------------------------------------
__global__ void vq_zero_losses(float* __restrict__ loss) {
    loss[0] = 0.0f;
    loss[1] = 0.0f;
}

// ---------------------------------------------------------------------------
// Kernel 5: gather codebook[idx] -> quantized [B,C,H,W]; accumulate MSE.
// ---------------------------------------------------------------------------
__global__ __launch_bounds__(256) void vq_gather(const float* __restrict__ x,
                                                 const float* __restrict__ cb,
                                                 const int* __restrict__ idx,
                                                 float* __restrict__ out,
                                                 float* __restrict__ loss) {
    const int b = blockIdx.x >> 8;
    const int c = blockIdx.x & 255;
    float ss = 0.0f;
#pragma unroll
    for (int i = 0; i < 4; ++i) {
        const int p = threadIdx.x + i * 256;
        const int id = idx[b * 1024 + p];
        const float q = cb[(size_t)id * CDIM + c];
        const size_t o = ((size_t)b * CDIM + c) * 1024 + p;
        const float f = x[o];
        out[o] = q;
        const float d = q - f;
        ss = fmaf(d, d, ss);
    }
    __shared__ float red[256];
    red[threadIdx.x] = ss;
    __syncthreads();
#pragma unroll
    for (int s2 = 128; s2 > 0; s2 >>= 1) {
        if (threadIdx.x < s2) red[threadIdx.x] += red[threadIdx.x + s2];
        __syncthreads();
    }
    if (threadIdx.x == 0) {
        const float v = red[0] * (1.0f / ((float)NROW * (float)CDIM));
        __hip_atomic_fetch_add(loss + 0, v, __ATOMIC_RELAXED, __HIP_MEMORY_SCOPE_AGENT);
        __hip_atomic_fetch_add(loss + 1, v, __ATOMIC_RELAXED, __HIP_MEMORY_SCOPE_AGENT);
    }
}

// ---------------------------------------------------------------------------
extern "C" void kernel_launch(void* const* d_in, const int* in_sizes, int n_in,
                              void* d_out, int out_size, void* d_ws, size_t ws_size,
                              hipStream_t stream) {
    const float* x  = (const float*)d_in[0];   // [16,256,32,32]
    const float* cb = (const float*)d_in[1];   // [8192,256]
    float* out      = (float*)d_out;

    char* ws = (char*)d_ws;                              // ~24.2 MB used
    __bf16* cbhi = (__bf16*)ws;
    __bf16* cblo = cbhi + (size_t)KCB * CDIM;
    __bf16* xhi  = cblo + (size_t)KCB * CDIM;
    __bf16* xlo  = xhi  + (size_t)NROW * CDIM;
    float*  e2   = (float*)(xlo + (size_t)NROW * CDIM);
    int*    idx  = (int*)(e2 + KCB);

    float* losses = out + (size_t)NROW * CDIM;

    vq_cb_prep<<<KCB / 8, 256, 0, stream>>>(cb, cbhi, cblo, e2);
    vq_x_prep<<<dim3(BB * HH, CDIM / 32), dim3(32, 8), 0, stream>>>(x, xhi, xlo);
    vq_argmin<<<NROW / 64, 128, 0, stream>>>(xhi, xlo, cbhi, cblo, e2, idx);
    vq_zero_losses<<<1, 1, 0, stream>>>(losses);
    vq_gather<<<BB * CDIM, 256, 0, stream>>>(x, cb, idx, out, losses);
}
